// DiscreteLatentDistributionVQ_76166950027350
// MI455X (gfx1250) — matrix operand order repaired and checked by reading, not compiled, not run on hardware
//
#include <hip/hip_runtime.h>
#include <hip/hip_bf16.h>

// ---------------------------------------------------------------------------
// VQ-VAE forward for MI455X (gfx1250, wave32, WMMA bf16 w/ fp32 accumulate)
// Round 2: 2-wide register blocking in the GEMM inner loops (A-fragment reuse
// across two B tiles -> half the ds_load traffic per v_wmma), plus
// global_prefetch_b8 on the streamed codebook tiles.
// ---------------------------------------------------------------------------

typedef __attribute__((ext_vector_type(16))) __bf16 v16bf;
typedef __attribute__((ext_vector_type(8)))  float  v8f;

#define NROWS 65536
#define INSZ  1024
#define HID   128
#define FEAT  512
#define KLAT  1024

__device__ __forceinline__ v8f wmma_bf16(v16bf a, v16bf b, v8f c) {
  // v_wmma_f32_16x16x32_bf16: (neg_a, A, neg_b, B, c_mod, C, reuse_a, reuse_b)
  return __builtin_amdgcn_wmma_f32_16x16x32_bf16(false, a, false, b, (short)0, c,
                                                 false, false);
}

__device__ __forceinline__ unsigned fkey(float v) {
  // order-preserving float -> uint mapping
  unsigned b = __float_as_uint(v);
  return (b & 0x80000000u) ? ~b : (b | 0x80000000u);
}

__device__ __forceinline__ unsigned wang(unsigned s) {
  s = (s ^ 61u) ^ (s >> 16);
  s *= 9u;
  s ^= s >> 4;
  s *= 0x27d4eb2du;
  s ^= s >> 15;
  return s;
}

// ---------------------------------------------------------------------------
// prep kernels
// ---------------------------------------------------------------------------
__global__ void zero_stats_kernel(unsigned* __restrict__ hist,
                                  float* __restrict__ sumsq,
                                  unsigned* __restrict__ maskcnt) {
  int t = blockIdx.x * blockDim.x + threadIdx.x;
  if (t < KLAT) hist[t] = 0u;
  if (t == 0) { *sumsq = 0.0f; *maskcnt = 0u; }
}

__global__ void f2bf_kernel(const float* __restrict__ src,
                            __bf16* __restrict__ dst, int n) {
  int i = blockIdx.x * blockDim.x + threadIdx.x;
  if (i < n) dst[i] = (__bf16)src[i];
}

__global__ void cbnorm_kernel(const float* __restrict__ cb,
                              float* __restrict__ cbn) {
  int k = blockIdx.x * blockDim.x + threadIdx.x;
  if (k < KLAT) {
    const float* row = cb + (size_t)k * FEAT;
    float s = 0.0f;
    for (int f = 0; f < FEAT; ++f) { float v = row[f]; s += v * v; }
    cbn[k] = s;
  }
}

// ---------------------------------------------------------------------------
// Kernel 1: z = (relu(X @ W1^T + b1)) @ W2^T + b2   (16 rows / workgroup)
// ---------------------------------------------------------------------------
__global__ __launch_bounds__(256)
void mlp_kernel(const float* __restrict__ X, const __bf16* __restrict__ W1bf,
                const float* __restrict__ b1, const __bf16* __restrict__ W2bf,
                const float* __restrict__ b2, float* __restrict__ zout,
                float* __restrict__ znorm) {
  __shared__ __bf16 Xs[16][INSZ + 8];
  __shared__ __bf16 Hs[16][HID + 8];
  __shared__ float  zn[16];

  const int tid  = threadIdx.x;
  const int wave = tid >> 5;
  const int lane = tid & 31;
  const int half = lane >> 4;
  const int l15  = lane & 15;
  const long long rowbase = (long long)blockIdx.x * 16;

  // stage X strip -> LDS as bf16
  for (int i = tid; i < 16 * INSZ; i += 256) {
    int r = i >> 10, c = i & (INSZ - 1);
    Xs[r][c] = (__bf16)X[(rowbase + r) * INSZ + c];
  }
  if (tid < 16) zn[tid] = 0.0f;
  __syncthreads();

  // ---- stage 1: H = relu(X W1^T + b1); each wave owns 16 hidden cols ----
  {
    const int colh = wave * 16 + l15;           // hidden unit per B-lane
    v8f acc = {0.f, 0.f, 0.f, 0.f, 0.f, 0.f, 0.f, 0.f};
    const __bf16* wr = W1bf + (size_t)colh * INSZ + half * 16;
    for (int kk = 0; kk < INSZ; kk += 32) {
      v16bf a, b;
      const int ka = kk + half * 16;
#pragma unroll
      for (int e = 0; e < 8; ++e) a[e] = Xs[l15][ka + e];
#pragma unroll
      for (int e = 0; e < 8; ++e) a[8 + e] = Xs[l15][ka + 8 + e];
#pragma unroll
      for (int e = 0; e < 16; ++e) b[e] = wr[kk + e];
      acc = wmma_bf16(a, b, acc);
    }
    const float bias = b1[colh];
#pragma unroll
    for (int r = 0; r < 8; ++r) {
      float v = acc[r] + bias;
      v = v > 0.f ? v : 0.f;
      Hs[r + 8 * half][colh] = (__bf16)v;       // C layout: m = r + 8*half
    }
  }
  __syncthreads();

  // ---- stage 2: Z = H W2^T + b2; each wave owns 4x16 output cols,
  //      processed as 2 passes of 2 tiles sharing the A fragment ----
  float np[8] = {0, 0, 0, 0, 0, 0, 0, 0};
  for (int tp = 0; tp < 2; ++tp) {
    const int colf0 = (wave * 4 + tp * 2) * 16 + l15;
    const int colf1 = colf0 + 16;
    v8f acc0 = {0.f, 0.f, 0.f, 0.f, 0.f, 0.f, 0.f, 0.f};
    v8f acc1 = {0.f, 0.f, 0.f, 0.f, 0.f, 0.f, 0.f, 0.f};
    const __bf16* wr0 = W2bf + (size_t)colf0 * HID + half * 16;
    const __bf16* wr1 = W2bf + (size_t)colf1 * HID + half * 16;
    for (int kk = 0; kk < HID; kk += 32) {
      v16bf a, b0, b1v;
      const int ka = kk + half * 16;
#pragma unroll
      for (int e = 0; e < 8; ++e) a[e] = Hs[l15][ka + e];
#pragma unroll
      for (int e = 0; e < 8; ++e) a[8 + e] = Hs[l15][ka + 8 + e];
#pragma unroll
      for (int e = 0; e < 16; ++e) b0[e] = wr0[kk + e];
#pragma unroll
      for (int e = 0; e < 16; ++e) b1v[e] = wr1[kk + e];
      acc0 = wmma_bf16(a, b0, acc0);
      acc1 = wmma_bf16(a, b1v, acc1);
    }
    const float bias0 = b2[colf0];
    const float bias1 = b2[colf1];
#pragma unroll
    for (int r = 0; r < 8; ++r) {
      const int m = r + 8 * half;
      float v0 = acc0[r] + bias0;
      float v1 = acc1[r] + bias1;
      zout[(rowbase + m) * FEAT + colf0] = v0;
      zout[(rowbase + m) * FEAT + colf1] = v1;
      np[r] += v0 * v0 + v1 * v1;
    }
  }
#pragma unroll
  for (int r = 0; r < 8; ++r) atomicAdd(&zn[r + 8 * half], np[r]);
  __syncthreads();
  if (tid < 16) znorm[rowbase + tid] = zn[tid];
}

// ---------------------------------------------------------------------------
// Kernel 2: distances, gumbel-categorical sample, encodings, quantized, loss
// ---------------------------------------------------------------------------
__global__ __launch_bounds__(256)
void vq_kernel(const float* __restrict__ z, const float* __restrict__ znorm,
               const __bf16* __restrict__ cbbf, const float* __restrict__ cbn,
               const float* __restrict__ cb, const int* __restrict__ mask,
               float* __restrict__ out_q, float* __restrict__ out_enc,
               float* __restrict__ out_nd, unsigned* __restrict__ hist,
               float* __restrict__ sumsq, unsigned* __restrict__ maskcnt) {
  __shared__ __bf16 Zs[16][FEAT + 8];
  __shared__ unsigned long long best[16];
  __shared__ float zn_s[16];
  __shared__ int   idx_s[16];
  __shared__ int   m_s[16];
  __shared__ float wsum[8];

  const int tid  = threadIdx.x;
  const int wave = tid >> 5;
  const int lane = tid & 31;
  const int half = lane >> 4;
  const int l15  = lane & 15;
  const long long rowbase = (long long)blockIdx.x * 16;

  for (int i = tid; i < 16 * FEAT; i += 256) {
    int r = i >> 9, c = i & (FEAT - 1);
    Zs[r][c] = (__bf16)z[(rowbase + r) * FEAT + c];
  }
  if (tid < 16) {
    zn_s[tid] = znorm[rowbase + tid];
    best[tid] = 0ull;
    m_s[tid]  = mask[rowbase + tid];
  }
  __syncthreads();

  // each wave covers 8 tiles of 16 codes (all 1024 codes across 8 waves),
  // processed as 4 passes of 2 tiles sharing the A fragment.
  unsigned long long bk[8] = {0ull, 0ull, 0ull, 0ull, 0ull, 0ull, 0ull, 0ull};
  for (int tp = 0; tp < 4; ++tp) {
    const int colk0 = (wave * 8 + tp * 2) * 16 + l15;   // codebook index / lane
    const int colk1 = colk0 + 16;
    const __bf16* cr0 = cbbf + (size_t)colk0 * FEAT + half * 16;
    const __bf16* cr1 = cbbf + (size_t)colk1 * FEAT + half * 16;
    // prefetch the next tile-pair's codebook rows (global_prefetch_b8)
    if (tp < 3) {
      __builtin_prefetch(cr0 + 32 * FEAT, 0, 0);
      __builtin_prefetch(cr1 + 32 * FEAT, 0, 0);
    }
    v8f acc0 = {0.f, 0.f, 0.f, 0.f, 0.f, 0.f, 0.f, 0.f};
    v8f acc1 = {0.f, 0.f, 0.f, 0.f, 0.f, 0.f, 0.f, 0.f};
    for (int ff = 0; ff < FEAT; ff += 32) {
      v16bf a, b0, b1v;
      const int ka = ff + half * 16;
#pragma unroll
      for (int e = 0; e < 8; ++e) a[e] = Zs[l15][ka + e];
#pragma unroll
      for (int e = 0; e < 8; ++e) a[8 + e] = Zs[l15][ka + 8 + e];
#pragma unroll
      for (int e = 0; e < 16; ++e) b0[e] = cr0[ff + e];
#pragma unroll
      for (int e = 0; e < 16; ++e) b1v[e] = cr1[ff + e];
      acc0 = wmma_bf16(a, b0, acc0);
      acc1 = wmma_bf16(a, b1v, acc1);
    }
    const float cn0 = cbn[colk0];
    const float cn1 = cbn[colk1];
#pragma unroll
    for (int r = 0; r < 8; ++r) {
      const int m = r + 8 * half;
      // tile 0
      {
        const float negd = 2.0f * acc0[r] - zn_s[m] - cn0;  // -||z - cb||^2
        out_nd[(rowbase + m) * KLAT + colk0] = negd;
        unsigned h = wang((unsigned)((rowbase + m) * KLAT + colk0) ^ 0x9E3779B9u);
        float u = ((float)(h >> 8) + 0.5f) * (1.0f / 16777216.0f);
        float g = -__logf(-__logf(u));
        unsigned long long key =
            ((unsigned long long)fkey(negd * 10.0f + g) << 32) | (unsigned)colk0;
        if (key > bk[r]) bk[r] = key;
      }
      // tile 1
      {
        const float negd = 2.0f * acc1[r] - zn_s[m] - cn1;
        out_nd[(rowbase + m) * KLAT + colk1] = negd;
        unsigned h = wang((unsigned)((rowbase + m) * KLAT + colk1) ^ 0x9E3779B9u);
        float u = ((float)(h >> 8) + 0.5f) * (1.0f / 16777216.0f);
        float g = -__logf(-__logf(u));
        unsigned long long key =
            ((unsigned long long)fkey(negd * 10.0f + g) << 32) | (unsigned)colk1;
        if (key > bk[r]) bk[r] = key;
      }
    }
  }
#pragma unroll
  for (int r = 0; r < 8; ++r) atomicMax(&best[r + 8 * half], bk[r]);
  __syncthreads();
  if (tid < 16) idx_s[tid] = (int)(best[tid] & 0xFFFFFFFFull);
  __syncthreads();

  // one-hot encodings
  for (int i = tid; i < 16 * KLAT; i += 256) {
    int r = i >> 10, c = i & (KLAT - 1);
    out_enc[(rowbase + r) * KLAT + c] = (c == idx_s[r]) ? 1.0f : 0.0f;
  }
  // quantized (== quantized_ste forward value) + masked loss partials
  float lsum = 0.0f;
  for (int i = tid; i < 16 * FEAT; i += 256) {
    int r = i >> 9, c = i & (FEAT - 1);
    float q = cb[(size_t)idx_s[r] * FEAT + c];
    out_q[(rowbase + r) * FEAT + c] = q;
    if (m_s[r] != 0) {
      float d = q - z[(rowbase + r) * FEAT + c];
      lsum += d * d;
    }
  }
#pragma unroll
  for (int o = 16; o > 0; o >>= 1) lsum += __shfl_down(lsum, o, 32);
  if (lane == 0) wsum[wave] = lsum;
  __syncthreads();
  if (tid == 0) {
    float s = 0.0f;
    for (int w = 0; w < 8; ++w) s += wsum[w];
    atomicAdd(sumsq, s);
  }
  if (tid < 16 && m_s[tid] != 0) {
    atomicAdd(&hist[idx_s[tid]], 1u);
    atomicAdd(maskcnt, 1u);
  }
}

// ---------------------------------------------------------------------------
// Kernel 3: scalars (loss, perplexity)
// ---------------------------------------------------------------------------
__global__ __launch_bounds__(1024)
void finalize_kernel(const unsigned* __restrict__ hist,
                     const float* __restrict__ sumsq,
                     const unsigned* __restrict__ maskcnt,
                     float* __restrict__ out_loss,
                     float* __restrict__ out_perp) {
  __shared__ float part[1024];
  const int t = threadIdx.x;
  const float M = (float)(*maskcnt);
  const float Minv = (M > 0.0f) ? (1.0f / M) : 0.0f;
  const float p = (float)hist[t] * Minv;
  part[t] = p * __logf(p + 1e-10f);
  __syncthreads();
  for (int s = 512; s > 0; s >>= 1) {
    if (t < s) part[t] += part[t + s];
    __syncthreads();
  }
  if (t == 0) {
    *out_perp = __expf(-part[0]);
    // q_latent_loss == e_latent_loss in the forward pass -> loss = 2 * MSE
    *out_loss = 2.0f * (*sumsq) / (M * (float)FEAT + 1e-20f);
  }
}

// ---------------------------------------------------------------------------
// launch
// ---------------------------------------------------------------------------
extern "C" void kernel_launch(void* const* d_in, const int* in_sizes, int n_in,
                              void* d_out, int out_size, void* d_ws,
                              size_t ws_size, hipStream_t stream) {
  const float* X    = (const float*)d_in[0];
  const int*   mask = (const int*)d_in[1];
  const float* W1   = (const float*)d_in[2];
  const float* b1   = (const float*)d_in[3];
  const float* W2   = (const float*)d_in[4];
  const float* b2   = (const float*)d_in[5];
  const float* cb   = (const float*)d_in[6];
  float* out = (float*)d_out;

  char* ws = (char*)d_ws;
  size_t off = 0;
  float*    z       = (float*)(ws + off); off += (size_t)NROWS * FEAT * 4;
  float*    znorm   = (float*)(ws + off); off += (size_t)NROWS * 4;
  __bf16*   W1bf    = (__bf16*)(ws + off); off += (size_t)HID * INSZ * 2;
  __bf16*   W2bf    = (__bf16*)(ws + off); off += (size_t)FEAT * HID * 2;
  __bf16*   cbbf    = (__bf16*)(ws + off); off += (size_t)KLAT * FEAT * 2;
  float*    cbn     = (float*)(ws + off); off += (size_t)KLAT * 4;
  unsigned* hist    = (unsigned*)(ws + off); off += (size_t)KLAT * 4;
  float*    sumsq   = (float*)(ws + off); off += 4;
  unsigned* maskcnt = (unsigned*)(ws + off); off += 4;

  // output layout: [loss][quantized_ste N*512][perplexity][encodings N*1024][-distances N*1024]
  float* out_loss = out;
  float* out_q    = out + 1;
  float* out_perp = out + (1 + (size_t)NROWS * FEAT);
  float* out_enc  = out + (2 + (size_t)NROWS * FEAT);
  float* out_nd   = out_enc + (size_t)NROWS * KLAT;

  zero_stats_kernel<<<(KLAT + 255) / 256, 256, 0, stream>>>(hist, sumsq, maskcnt);
  f2bf_kernel<<<(HID * INSZ + 255) / 256, 256, 0, stream>>>(W1, W1bf, HID * INSZ);
  f2bf_kernel<<<(FEAT * HID + 255) / 256, 256, 0, stream>>>(W2, W2bf, FEAT * HID);
  f2bf_kernel<<<(KLAT * FEAT + 255) / 256, 256, 0, stream>>>(cb, cbbf, KLAT * FEAT);
  cbnorm_kernel<<<KLAT / 256, 256, 0, stream>>>(cb, cbn);
  mlp_kernel<<<NROWS / 16, 256, 0, stream>>>(X, W1bf, b1, W2bf, b2, z, znorm);
  vq_kernel<<<NROWS / 16, 256, 0, stream>>>(z, znorm, cbbf, cbn, cb, mask, out_q,
                                            out_enc, out_nd, hist, sumsq, maskcnt);
  finalize_kernel<<<1, 1024, 0, stream>>>(hist, sumsq, maskcnt, out_loss, out_perp);
}
